// MultiHeadDotProductAttention_89146341196502
// MI455X (gfx1250) — matrix operand order repaired and checked by reading, not compiled, not run on hardware
//
#include <hip/hip_runtime.h>

// ---------------------------------------------------------------------------
// Sliding-window MHA for MI455X (gfx1250, wave32, WMMA bf16 -> f32 accum).
//   B=4, T=2048, E=1024, H=16 heads x 64 dim, band |i-j| <= 256.
// ~86 GFLOP vs ~120 MB of HBM traffic (~5us at 23.3 TB/s) -> compute-bound,
// so all four GEMM stages run on v_wmma_f32_16x16x32_bf16. K/Q tiles are
// staged by the Tensor Data Mover (TENSOR_LOAD_TO_LDS + s_wait_tensorcnt);
// all row-crossing WMMA B-operands use a pair-interleaved dword LDS layout
// so every fragment is two ds_load_b128 instead of 16 ds_load_u16 + packs.
// ---------------------------------------------------------------------------

#define TSEQ  2048
#define EMB   1024
#define NHEAD 16
#define HDIM  64
#define MROWS 8192   // B*T
#define WIN   256    // band half-width

typedef __attribute__((ext_vector_type(16))) __bf16       v16bf;
typedef __attribute__((ext_vector_type(8)))  float        v8f;
typedef __attribute__((ext_vector_type(8)))  unsigned int v8u;
typedef __attribute__((ext_vector_type(4)))  unsigned int u32x4;
typedef __attribute__((ext_vector_type(8)))  int          i32x8;
typedef __attribute__((ext_vector_type(4)))  int          i32x4;

union Frag { v8u u; v16bf h; };

#if defined(__HIP_DEVICE_COMPILE__) && __has_builtin(__builtin_amdgcn_tensor_load_to_lds)
#define USE_TDM 1
#else
#define USE_TDM 0
#endif

__device__ __forceinline__ unsigned short f2bf(float f) {
  unsigned u = __float_as_uint(f);
  u += 0x7FFFu + ((u >> 16) & 1u);          // round-to-nearest-even
  return (unsigned short)(u >> 16);
}
__device__ __forceinline__ unsigned pk2(float a, float b) {
  return (unsigned)f2bf(a) | ((unsigned)f2bf(b) << 16);
}

__device__ __forceinline__ v8f wmma_bf16(const Frag& a, const Frag& b, v8f c) {
#if defined(__HIP_DEVICE_COMPILE__)
  return __builtin_amdgcn_wmma_f32_16x16x32_bf16(
      /*neg_a=*/false, a.h, /*neg_b=*/false, b.h,
      /*c_mod=*/(short)0, c, /*reuse_a=*/false, /*reuse_b=*/false);
#else
  return c;
#endif
}

__device__ __forceinline__ float rmax16(float v) {
#pragma unroll
  for (int off = 1; off < 16; off <<= 1) v = fmaxf(v, __shfl_xor(v, off, 32));
  return v;
}
__device__ __forceinline__ float rsum16(float v) {
#pragma unroll
  for (int off = 1; off < 16; off <<= 1) v += __shfl_xor(v, off, 32);
  return v;
}

#if USE_TDM
// TDM load of a 2D bf16 tile (row length 64 elements) out of a [2048 x 64]
// tensor into LDS, padding each 128B row by 16B -> LDS row stride 72 halves.
// D# layout per CDNA5 ISA (group0 128b, group1 256b).
// This toolchain's builtin takes 6 args (4 SGPR groups + spare group + cpol).
__device__ __forceinline__ void tdm_load_2d(unsigned lds_off, const void* g,
                                            unsigned rows) {
  unsigned long long ga = (unsigned long long)(size_t)g;
  u32x4 g0 = {1u,                               // count=1, user mode
              lds_off,                          // lds_addr (bytes)
              (unsigned)ga,                     // global_addr[31:0]
              (unsigned)((ga >> 32) & 0x01FFFFFFull) | (2u << 30)};  // type=2
  i32x8 g1 = {(int)((1u << 16)                  // data_size = 2 bytes
                    | (1u << 20)                // pad_enable
                    | (4u << 22)                // pad_interval: 32 DWORDs
                    | (3u << 25)),              // pad_amount: 4 DWORDs (16B)
              (int)(64u << 16),                 // tensor_dim0 = 64
              (int)(2048u << 16),               // tensor_dim1 = 2048
              (int)(64u << 16),                 // tile_dim0 = 64
              (int)rows,                        // tile_dim1 = rows
              64,                               // tensor_dim0_stride = 64
              0, 0};
  i32x4 z4 = {0, 0, 0, 0};
  i32x8 z8 = {0, 0, 0, 0, 0, 0, 0, 0};
  __builtin_amdgcn_tensor_load_to_lds(g0, g1, z4, z4, z8, 0);
}
#endif

// ---------------------------------------------------------------------------
// Kernel 1: fused QKV projection + rotary + softmax scale (on Q).
// blockIdx.z selects {Q, K, V}. Output layout: bf16 [B, H, T, 64].
// Weight tile in LDS is pair-interleaved: Bsd[n][k/2] dword = (W[k][n],W[k+1][n]).
// ---------------------------------------------------------------------------
__global__ __launch_bounds__(256) void k_qkv_rope(
    const float* __restrict__ Xq, const float* __restrict__ Xkv,
    const float* __restrict__ Wq, const float* __restrict__ Wk,
    const float* __restrict__ Wv,
    unsigned short* __restrict__ Qo, unsigned short* __restrict__ Ko,
    unsigned short* __restrict__ Vo) {
  __shared__ unsigned short As[128 * 40];   // 128 x 32 bf16, stride 40
  __shared__ unsigned int   Bsd[128 * 20];  // [n][k/2] pair dwords, stride 20

  const int which = blockIdx.z;
  const float* A = (which == 0) ? Xq : Xkv;
  const float* W = (which == 0) ? Wq : ((which == 1) ? Wk : Wv);
  unsigned short* D = (which == 0) ? Qo : ((which == 1) ? Ko : Vo);
  const bool  rope  = (which < 2);
  const float scale = (which == 0) ? 0.125f : 1.0f;   // 1/sqrt(64) folded into Q

  const int rowBase = blockIdx.y * 128;
  const int colBase = blockIdx.x * 128;
  const int t = threadIdx.x;
  const int wid = t >> 5, lane = t & 31, lh = lane >> 4, ln = lane & 15;
  const int wm = (wid >> 2) * 64, wn = (wid & 3) * 32;   // 2x4 wave grid

  v8f acc[4][2];
#pragma unroll
  for (int mt = 0; mt < 4; ++mt)
#pragma unroll
    for (int nt = 0; nt < 2; ++nt) acc[mt][nt] = v8f{0, 0, 0, 0, 0, 0, 0, 0};

  for (int k0 = 0; k0 < EMB; k0 += 32) {
    {  // stage A tile: 128x32 f32 -> bf16 (2 threads / row)
      const int r = t >> 1, cb = (t & 1) * 16;
      const float4* ap = (const float4*)(A + (size_t)(rowBase + r) * EMB + k0 + cb);
      unsigned* dp = (unsigned*)&As[r * 40 + cb];
      if (k0 + 32 < EMB) __builtin_prefetch(ap + 8, 0, 1);  // global_prefetch_b8
#pragma unroll
      for (int j = 0; j < 4; ++j) {
        float4 f = ap[j];
        dp[2 * j]     = pk2(f.x, f.y);
        dp[2 * j + 1] = pk2(f.z, f.w);
      }
    }
    {  // stage W tile pair-interleaved: thread owns k-row pair 2p,2p+1 x 8 cols
      const int p = t >> 4, c0 = (t & 15) * 8;
      const float* we = W + (size_t)(k0 + 2 * p) * EMB + colBase + c0;
      if (k0 + 32 < EMB) __builtin_prefetch(we + (size_t)32 * EMB, 0, 1);
      float4 e0 = ((const float4*)we)[0], e1 = ((const float4*)we)[1];
      float4 o0 = ((const float4*)(we + EMB))[0], o1 = ((const float4*)(we + EMB))[1];
      Bsd[(c0 + 0) * 20 + p] = pk2(e0.x, o0.x);
      Bsd[(c0 + 1) * 20 + p] = pk2(e0.y, o0.y);
      Bsd[(c0 + 2) * 20 + p] = pk2(e0.z, o0.z);
      Bsd[(c0 + 3) * 20 + p] = pk2(e0.w, o0.w);
      Bsd[(c0 + 4) * 20 + p] = pk2(e1.x, o1.x);
      Bsd[(c0 + 5) * 20 + p] = pk2(e1.y, o1.y);
      Bsd[(c0 + 6) * 20 + p] = pk2(e1.z, o1.z);
      Bsd[(c0 + 7) * 20 + p] = pk2(e1.w, o1.w);
    }
    __syncthreads();

    Frag af[4];
#pragma unroll
    for (int mt = 0; mt < 4; ++mt) {       // A frag: M=ln, K = 16*lh + s (contiguous)
      const int r = wm + mt * 16 + ln;
#pragma unroll
      for (int i = 0; i < 8; ++i) {
        const int kk = 2 * i + 16 * lh;
        af[mt].u[i] = *(const unsigned*)&As[r * 40 + kk];
      }
    }
#pragma unroll
    for (int nt = 0; nt < 2; ++nt) {       // B frag: 8 consecutive pair dwords
      Frag bfrag;
      const int c = wn + nt * 16 + ln;
#pragma unroll
      for (int i = 0; i < 8; ++i) bfrag.u[i] = Bsd[c * 20 + 8 * lh + i];
#pragma unroll
      for (int mt = 0; mt < 4; ++mt)
        acc[mt][nt] = wmma_bf16(af[mt], bfrag, acc[mt][nt]);
    }
    __syncthreads();
  }

  // Epilogue: scale, rotary (pair partner lives in lane^1), store [B,H,T,64].
#pragma unroll
  for (int mt = 0; mt < 4; ++mt)
#pragma unroll
    for (int nt = 0; nt < 2; ++nt)
#pragma unroll
      for (int e = 0; e < 8; ++e) {
        const int M = rowBase + wm + mt * 16 + e + 8 * lh;   // token row
        const int N = colBase + wn + nt * 16 + ln;           // feature col
        float v = acc[mt][nt][e] * scale;
        if (rope) {
          const int   pos = M & (TSEQ - 1);
          const int   i2  = (N & 63) >> 1;
          // inv_freq = 10000^(-i/32) = exp(-i * ln(1e4)/32)
          const float fr  = __expf(-(float)i2 * (9.210340371976184f / 32.0f));
          float c, s;
          __sincosf((float)pos * fr, &s, &c);
          const float partner = __shfl_xor(v, 1, 32);
          const float rot = (N & 1) ? partner : -partner;
          v = v * c + rot * s;
        }
        const int b  = M >> 11, tt = M & (TSEQ - 1);
        const int h  = N >> 6,  d  = N & 63;
        D[((size_t)(b * NHEAD + h) * TSEQ + tt) * HDIM + d] = f2bf(v);
      }
}

// ---------------------------------------------------------------------------
// Kernel 2: banded flash attention. grid = (T/128 query tiles, B*H).
// Each wave: 16 query rows x full 64-key tile -> wave-local softmax stats.
// Q/K tiles DMA'd by the Tensor Data Mover; V pair-interleaved by VALU.
// ---------------------------------------------------------------------------
__global__ __launch_bounds__(256) void k_attn(
    const unsigned short* __restrict__ Qb, const unsigned short* __restrict__ Kb,
    const unsigned short* __restrict__ Vb, unsigned short* __restrict__ AO) {
  __shared__ unsigned short Qs[128 * 72];      // 18.0 KB (TDM, padded rows)
  __shared__ unsigned short Ks[64 * 72];       //  9.0 KB (TDM, padded rows)
  __shared__ unsigned int   Vsd[64 * 20];      //  5.0 KB [dim][tokenpair]
  __shared__ unsigned short Ps[8 * 16 * 72];   // 18.0 KB (per-wave P strips)

  const int qBase = blockIdx.x * 128;
  const int bh    = blockIdx.y;
  const unsigned short* Qp = Qb + (size_t)bh * TSEQ * HDIM;
  const unsigned short* Kp = Kb + (size_t)bh * TSEQ * HDIM;
  const unsigned short* Vp = Vb + (size_t)bh * TSEQ * HDIM;

  const int t = threadIdx.x;
  const int wid = t >> 5, lane = t & 31, lh = lane >> 4, ln = lane & 15;

#if USE_TDM
  if (wid == 0) {
    tdm_load_2d((unsigned)(size_t)&Qs[0], Qp + (size_t)qBase * HDIM, 128u);
    __builtin_amdgcn_s_wait_tensorcnt(0);
  }
#else
  {  // fallback: manual Q stage 128x64 bf16
    const int r = t >> 1, cb = (t & 1) * 32;
    const uint4* src = (const uint4*)(Qp + (size_t)(qBase + r) * HDIM + cb);
    uint4* dst = (uint4*)&Qs[r * 72 + cb];
#pragma unroll
    for (int j = 0; j < 4; ++j) dst[j] = src[j];
  }
#endif
  __syncthreads();

  Frag qf[2];
#pragma unroll
  for (int ks = 0; ks < 2; ++ks)
#pragma unroll
    for (int i = 0; i < 8; ++i) {
      const int kk = ks * 32 + ((i < 4) ? 2 * i : 16 + 2 * (i - 4)) + 8 * lh;
      qf[ks].u[i] = *(const unsigned*)&Qs[(wid * 16 + ln) * 72 + kk];
    }

  v8f o[4];
#pragma unroll
  for (int nt = 0; nt < 4; ++nt) o[nt] = v8f{0, 0, 0, 0, 0, 0, 0, 0};
  float rowmax[8], rowsum[8];
#pragma unroll
  for (int e = 0; e < 8; ++e) { rowmax[e] = -__builtin_inff(); rowsum[e] = 0.f; }

  const int jStart = (qBase - WIN > 0) ? (qBase - WIN) : 0;
  const int jEndV  = qBase + 128 + WIN;
  const int jEnd   = (jEndV < TSEQ) ? jEndV : TSEQ;

  for (int j0 = jStart; j0 < jEnd; j0 += 64) {
    __syncthreads();
#if USE_TDM
    if (wid == 0)  // kick K DMA, then overlap V staging on all waves
      tdm_load_2d((unsigned)(size_t)&Ks[0], Kp + (size_t)j0 * HDIM, 64u);
#else
    {  // fallback: manual K stage 64x64 bf16
      const int r = t >> 2, cb = (t & 3) * 16;
      const uint4* src = (const uint4*)(Kp + (size_t)(j0 + r) * HDIM + cb);
      uint4* dst = (uint4*)&Ks[r * 72 + cb];
      dst[0] = src[0]; dst[1] = src[1];
    }
#endif
    {  // V stage, pair-interleaved: Vsd[dim][p] = (V[2p][dim], V[2p+1][dim])
      const int p = t >> 3, d0 = (t & 7) * 8;
      const unsigned short* v0 = Vp + (size_t)(j0 + 2 * p) * HDIM + d0;
      uint4 a = *(const uint4*)v0;
      uint4 b = *(const uint4*)(v0 + HDIM);
      const unsigned* au = (const unsigned*)&a;
      const unsigned* bu = (const unsigned*)&b;
#pragma unroll
      for (int jj = 0; jj < 8; ++jj) {
        const unsigned lo = (au[jj >> 1] >> (16 * (jj & 1))) & 0xFFFFu;
        const unsigned hi = (bu[jj >> 1] >> (16 * (jj & 1))) & 0xFFFFu;
        Vsd[(d0 + jj) * 20 + p] = lo | (hi << 16);
      }
    }
#if USE_TDM
    if (wid == 0) __builtin_amdgcn_s_wait_tensorcnt(0);
#endif
    __syncthreads();

    // S = Q * K^T  (16 rows x 64 key cols per wave, 4 N-tiles)
    v8f s[4];
#pragma unroll
    for (int nt = 0; nt < 4; ++nt) {
      s[nt] = v8f{0, 0, 0, 0, 0, 0, 0, 0};
#pragma unroll
      for (int ks = 0; ks < 2; ++ks) {
        Frag kb;                       // B[k][n] = K[token n][dim k]
#pragma unroll
        for (int i = 0; i < 8; ++i) {
          const int kk = ks * 32 + ((i < 4) ? 2 * i : 16 + 2 * (i - 4)) + 8 * lh;
          kb.u[i] = *(const unsigned*)&Ks[(nt * 16 + ln) * 72 + kk];
        }
        s[nt] = wmma_bf16(qf[ks], kb, s[nt]);
      }
    }

    // band mask + online softmax (row stats wave-local via 16-lane shfl)
#pragma unroll
    for (int e = 0; e < 8; ++e) {
      const int row = qBase + wid * 16 + e + 8 * lh;
      float tmax = -__builtin_inff();
#pragma unroll
      for (int nt = 0; nt < 4; ++nt) {
        const int dd = row - (j0 + nt * 16 + ln);
        if (dd > WIN || dd < -WIN) s[nt][e] = -__builtin_inff();
        tmax = fmaxf(tmax, s[nt][e]);
      }
      tmax = rmax16(tmax);
      const float mn   = fmaxf(rowmax[e], tmax);
      const float mn_s = (mn == -__builtin_inff()) ? 0.f : mn;  // safe subtrahend
      const float alpha = __expf(rowmax[e] - mn_s);
      rowmax[e] = mn;
      float psum = 0.f;
#pragma unroll
      for (int nt = 0; nt < 4; ++nt) {
        const float p = __expf(s[nt][e] - mn_s);
        s[nt][e] = p;
        psum += p;
      }
      rowsum[e] = rowsum[e] * alpha + rsum16(psum);
#pragma unroll
      for (int nt = 0; nt < 4; ++nt) o[nt][e] *= alpha;
    }

    // P: C-layout -> A-layout via per-wave LDS strip (bf16)
#pragma unroll
    for (int nt = 0; nt < 4; ++nt)
#pragma unroll
      for (int e = 0; e < 8; ++e)
        Ps[(wid * 16 + e + 8 * lh) * 72 + nt * 16 + ln] = f2bf(s[nt][e]);
    __syncthreads();

    // O += P * V   (V frag = 8 consecutive pair dwords)
#pragma unroll
    for (int ks = 0; ks < 2; ++ks) {
      Frag pa;
#pragma unroll
      for (int i = 0; i < 8; ++i) {
        const int kk = ks * 32 + 2 * i + 16 * lh;
        pa.u[i] = *(const unsigned*)&Ps[(wid * 16 + ln) * 72 + kk];
      }
#pragma unroll
      for (int nt = 0; nt < 4; ++nt) {
        Frag vbf;                      // B[k][n] = V[token k][dim n]
        const int n = nt * 16 + ln;
#pragma unroll
        for (int i = 0; i < 8; ++i)
          vbf.u[i] = Vsd[n * 20 + ks * 16 + 8 * lh + i];
        o[nt] = wmma_bf16(pa, vbf, o[nt]);
      }
    }
  }

  // normalize + store to [B*T, H*D] bf16 for the output projection
  const int b = bh >> 4, h = bh & 15;
#pragma unroll
  for (int nt = 0; nt < 4; ++nt)
#pragma unroll
    for (int e = 0; e < 8; ++e) {
      const float val = o[nt][e] / rowsum[e];
      const int M = qBase + wid * 16 + e + 8 * lh;
      const size_t row_g = (size_t)b * TSEQ + M;
      const int col = h * HDIM + nt * 16 + ln;
      AO[row_g * EMB + col] = f2bf(val);
    }
}

// ---------------------------------------------------------------------------
// Kernel 3: output projection: f32 out = bf16 AO[8192x1024] * Wo[1024x1024].
// ---------------------------------------------------------------------------
__global__ __launch_bounds__(256) void k_oproj(
    const unsigned short* __restrict__ AO, const float* __restrict__ Wo,
    float* __restrict__ Out) {
  __shared__ unsigned short As[128 * 40];
  __shared__ unsigned int   Bsd[128 * 20];

  const int rowBase = blockIdx.y * 128;
  const int colBase = blockIdx.x * 128;
  const int t = threadIdx.x;
  const int wid = t >> 5, lane = t & 31, lh = lane >> 4, ln = lane & 15;
  const int wm = (wid >> 2) * 64, wn = (wid & 3) * 32;

  v8f acc[4][2];
#pragma unroll
  for (int mt = 0; mt < 4; ++mt)
#pragma unroll
    for (int nt = 0; nt < 2; ++nt) acc[mt][nt] = v8f{0, 0, 0, 0, 0, 0, 0, 0};

  for (int k0 = 0; k0 < EMB; k0 += 32) {
    {  // stage A (already bf16): straight uint4 copies
      const int r = t >> 1, cb = (t & 1) * 16;
      const uint4* ap = (const uint4*)(AO + (size_t)(rowBase + r) * EMB + k0 + cb);
      uint4* dp = (uint4*)&As[r * 40 + cb];
      if (k0 + 32 < EMB) __builtin_prefetch(ap + 4, 0, 1);
      dp[0] = ap[0]; dp[1] = ap[1];
    }
    {  // stage Wo tile pair-interleaved (f32 -> bf16 pairs)
      const int p = t >> 4, c0 = (t & 15) * 8;
      const float* we = Wo + (size_t)(k0 + 2 * p) * EMB + colBase + c0;
      float4 e0 = ((const float4*)we)[0], e1 = ((const float4*)we)[1];
      float4 o0 = ((const float4*)(we + EMB))[0], o1 = ((const float4*)(we + EMB))[1];
      Bsd[(c0 + 0) * 20 + p] = pk2(e0.x, o0.x);
      Bsd[(c0 + 1) * 20 + p] = pk2(e0.y, o0.y);
      Bsd[(c0 + 2) * 20 + p] = pk2(e0.z, o0.z);
      Bsd[(c0 + 3) * 20 + p] = pk2(e0.w, o0.w);
      Bsd[(c0 + 4) * 20 + p] = pk2(e1.x, o1.x);
      Bsd[(c0 + 5) * 20 + p] = pk2(e1.y, o1.y);
      Bsd[(c0 + 6) * 20 + p] = pk2(e1.z, o1.z);
      Bsd[(c0 + 7) * 20 + p] = pk2(e1.w, o1.w);
    }
    __syncthreads();

    Frag af[4];
#pragma unroll
    for (int mt = 0; mt < 4; ++mt) {
      const int r = wm + mt * 16 + ln;
#pragma unroll
      for (int i = 0; i < 8; ++i) {
        const int kk = 2 * i + 16 * lh;
        af[mt].u[i] = *(const unsigned*)&As[r * 40 + kk];
      }
    }
#pragma unroll
    for (int nt = 0; nt < 2; ++nt) {
      Frag bfrag;
      const int c = wn + nt * 16 + ln;
#pragma unroll
      for (int i = 0; i < 8; ++i) bfrag.u[i] = Bsd[c * 20 + 8 * lh + i];
#pragma unroll
      for (int mt = 0; mt < 4; ++mt)
        acc[mt][nt] = wmma_bf16(af[mt], bfrag, acc[mt][nt]);
    }
    __syncthreads();
  }

#pragma unroll
  for (int mt = 0; mt < 4; ++mt)
#pragma unroll
    for (int nt = 0; nt < 2; ++nt)
#pragma unroll
      for (int e = 0; e < 8; ++e) {
        const int M = rowBase + wm + mt * 16 + e + 8 * lh;
        const int N = colBase + wn + nt * 16 + ln;
        Out[(size_t)M * EMB + N] = acc[mt][nt][e];
      }
}

// ---------------------------------------------------------------------------
extern "C" void kernel_launch(void* const* d_in, const int* in_sizes, int n_in,
                              void* d_out, int out_size, void* d_ws, size_t ws_size,
                              hipStream_t stream) {
  (void)in_sizes; (void)n_in; (void)out_size; (void)ws_size;
  const float* inq  = (const float*)d_in[0];
  const float* inkv = (const float*)d_in[1];
  const float* Wq   = (const float*)d_in[2];
  const float* Wk   = (const float*)d_in[3];
  const float* Wv   = (const float*)d_in[4];
  const float* Wo   = (const float*)d_in[5];

  // Workspace: Q, K, V bf16 [B,H,T,64] + attention output bf16 [B*T, H*D].
  const size_t PLANE = (size_t)MROWS * EMB;  // 8,388,608 elems (16 MiB bf16)
  unsigned short* ws = (unsigned short*)d_ws;
  unsigned short* Qb = ws;
  unsigned short* Kb = ws + PLANE;
  unsigned short* Vb = ws + 2 * PLANE;
  unsigned short* AO = ws + 3 * PLANE;

  k_qkv_rope<<<dim3(EMB / 128, MROWS / 128, 3), 256, 0, stream>>>(
      inq, inkv, Wq, Wk, Wv, Qb, Kb, Vb);
  k_attn<<<dim3(TSEQ / 128, 4 * NHEAD), 256, 0, stream>>>(Qb, Kb, Vb, AO);
  k_oproj<<<dim3(EMB / 128, MROWS / 128), 256, 0, stream>>>(AO, Wo, (float*)d_out);
}
